// ODECell_40956808135369
// MI455X (gfx1250) — compile-verified, matrix operand order checked by reading, not run on hardware
//
#include <hip/hip_runtime.h>

typedef __attribute__((ext_vector_type(16))) _Float16 v16h;
typedef __attribute__((ext_vector_type(8)))  float    v8f;

#define OMEGA_C 0.1f
#define DT_U    (0.1f / 6.0f)
#define BATCH_N 2048
#define UNITS_N 128
#define INDIM_N 128

__device__ __forceinline__ float f4get(float4 v, int j) {
  return j == 0 ? v.x : j == 1 ? v.y : j == 2 ? v.z : v.w;
}

// One wave handles a (1 batch) x (16 units) tile, reducing over D=128 with
// compensated-f16 WMMA row-sums (B = ones). A-operand built directly in the
// ISA 16x32 f16 layout: lane = m + 16h holds K = 8h+j and 16+8h+j per chunk.
__global__ __launch_bounds__(256)
void ode_cell_wmma_kernel(const float* __restrict__ inputs,
                          const float* __restrict__ state,
                          const float* __restrict__ A,
                          const float* __restrict__ sigma,
                          const float* __restrict__ mu,
                          float* __restrict__ out)
{
  const int lane = threadIdx.x & 31;
  const int wave = threadIdx.x >> 5;
  const int tile = blockIdx.x * 8 + wave;     // 2048 blocks * 8 waves = 16384 tiles
  const int b    = tile >> 3;                 // batch index
  const int u0   = (tile & 7) << 4;           // unit-tile base (0,16,...,112)
  const int m    = lane & 15;                 // M row within operand
  const int h    = lane >> 4;                 // lane half selects K sub-range

  const float4* __restrict__ xrow = (const float4*)(inputs + (size_t)b * INDIM_N);
  const float4* __restrict__ srow = (const float4*)(sigma  + (size_t)(u0 + m) * INDIM_N);
  const float4* __restrict__ mrow = (const float4*)(mu     + (size_t)(u0 + m) * INDIM_N);
  const float4* __restrict__ arow = (const float4*)(A      + (size_t)(u0 + m) * INDIM_N);

  v16h ones;
  #pragma unroll
  for (int i = 0; i < 16; ++i) ones[i] = (_Float16)1.0f;

  v8f c_fa = {};   // accumulates sum_d f * A  (rows = units)
  v8f c_fs = {};   // accumulates sum_d f

  #pragma unroll
  for (int c = 0; c < 4; ++c) {               // 4 chunks of K=32 cover D=128
    v16h fh, fl, gh, gl;
    #pragma unroll
    for (int sb = 0; sb < 2; ++sb) {
      // d = 32*c + 16*sb + 8*h  (+ j, j=0..7); float4 index q = d/4
      const int q = c * 8 + sb * 4 + h * 2;
      float4 x0 = xrow[q], x1 = xrow[q + 1];
      float4 s0 = srow[q], s1 = srow[q + 1];
      float4 m0 = mrow[q], m1 = mrow[q + 1];
      float4 a0 = arow[q], a1 = arow[q + 1];
      #pragma unroll
      for (int j = 0; j < 8; ++j) {
        float xv = (j < 4) ? f4get(x0, j) : f4get(x1, j - 4);
        float sv = (j < 4) ? f4get(s0, j) : f4get(s1, j - 4);
        float mv = (j < 4) ? f4get(m0, j) : f4get(m1, j - 4);
        float av = (j < 4) ? f4get(a0, j) : f4get(a1, j - 4);
        float t = sv * (xv - mv);
        float e = __expf(-t);                        // v_exp_f32 path
        float f = __builtin_amdgcn_rcpf(1.0f + e);   // sigmoid
        float g = f * av;
        // compensated split: value = hi + lo (both f16), f32-accurate sum
        _Float16 f_hi = (_Float16)f;
        _Float16 f_lo = (_Float16)(f - (float)f_hi);
        _Float16 g_hi = (_Float16)g;
        _Float16 g_lo = (_Float16)(g - (float)g_hi);
        const int idx = sb * 8 + j;                  // halves 0..7 then 8..15
        fh[idx] = f_hi; fl[idx] = f_lo;
        gh[idx] = g_hi; gl[idx] = g_lo;
      }
    }
    // Row-sum via WMMA against ones; accumulate hi and lo parts in f32 C.
    c_fs = __builtin_amdgcn_wmma_f32_16x16x32_f16(false, fh, false, ones, (short)0, c_fs, false, false);
    c_fs = __builtin_amdgcn_wmma_f32_16x16x32_f16(false, fl, false, ones, (short)0, c_fs, false, false);
    c_fa = __builtin_amdgcn_wmma_f32_16x16x32_f16(false, gh, false, ones, (short)0, c_fa, false, false);
    c_fa = __builtin_amdgcn_wmma_f32_16x16x32_f16(false, gl, false, ones, (short)0, c_fa, false, false);
  }

  // C/D layout: VGPR r holds row r (lanes 0-15) / row 8+r (lanes 16-31),
  // identical across columns. Lanes 0 and 16 finish 8 units each with the
  // 6-step Euler unfold and store.
  if (m == 0) {
    const int ubase = u0 + 8 * h;
    #pragma unroll
    for (int r = 0; r < 8; ++r) {
      float fa = c_fa[r];
      float fs = c_fs[r];
      float x  = state[(size_t)b * UNITS_N + ubase + r];
      #pragma unroll
      for (int it = 0; it < 6; ++it)
        x += DT_U * (-OMEGA_C * x + fa - x * fs);
      out[(size_t)b * UNITS_N + ubase + r] = x;
    }
  }
}

extern "C" void kernel_launch(void* const* d_in, const int* in_sizes, int n_in,
                              void* d_out, int out_size, void* d_ws, size_t ws_size,
                              hipStream_t stream) {
  const float* inputs = (const float*)d_in[0];
  const float* state  = (const float*)d_in[1];
  const float* A      = (const float*)d_in[2];
  const float* sigma  = (const float*)d_in[3];
  const float* mu     = (const float*)d_in[4];
  float* out = (float*)d_out;

  const int tiles  = BATCH_N * (UNITS_N / 16);   // 16384 wave-tiles
  const int blocks = tiles / 8;                  // 8 waves per 256-thread block
  hipLaunchKernelGGL(ode_cell_wmma_kernel, dim3(blocks), dim3(256), 0, stream,
                     inputs, state, A, sigma, mu, out);
}
